// TestModel_37692632990109
// MI455X (gfx1250) — matrix operand order repaired and checked by reading, not compiled, not run on hardware
//
#include <hip/hip_runtime.h>
#include <cmath>

// ---------------- problem constants (match reference) ----------------
constexpr int Tt = 1000;   // sequence length
constexpr int Bb = 1024;   // batch
constexpr int Oo = 919;    // fc out

typedef __attribute__((ext_vector_type(16))) __bf16 v16bf;
typedef __attribute__((ext_vector_type(8)))  float  v8f;

__device__ __forceinline__ float sigm(float x) { return 1.f / (1.f + __expf(-x)); }

// A-fragment K index for 16-bit 16x32 A matrix (contiguous order, matches B):
// lanes 0-15 hold K {0..15}, lanes 16-31 hold K {16..31}
__device__ __forceinline__ int kmapA(int j, int hi) {
    return j + hi * 16;
}

union PK2 { __bf16 h[2]; unsigned int u; };

// =====================================================================
// One bidirectional-LSTM layer. grid = (B/16, 2 dirs), block = 256 (8 waves).
// Each block owns 16 batch rows for one direction and runs all T steps
// privately (the recurrence is independent per batch row -> no global sync).
// Wave w computes gate columns [16w, 16w+16) of z = x@Wih^T + h@Whh^T + b.
// =====================================================================
template <int IN, bool WRITE_Y>
__global__ __launch_bounds__(256) void lstm_layer_kernel(
    const void*  __restrict__ xin,   // IN==4: fp32 [T][B][4]; else bf16 [T][B][64]
    const float* __restrict__ Wih,   // [2][128][IN]
    const float* __restrict__ Whh,   // [2][128][32]
    const float* __restrict__ bias,  // [2][128]
    __bf16*      __restrict__ yout,  // bf16 [T][B][64] (dir-interleaved) or null
    __bf16*      __restrict__ feat,  // bf16 [B][192] final hidden features
    int layer)
{
    __shared__ alignas(16) float  zs[16][132];   // gate pre-activations (padded)
    __shared__ alignas(16) __bf16 hs[16][40];    // recurrent h state (80B rows, 16B-aligned)
    __shared__ alignas(16) __bf16 xsb[16][72];   // staged input tile for IN==64 (144B rows)
    __shared__ alignas(16) float  xsf[16][4];    // staged input tile for IN==4

    const int tid  = threadIdx.x;
    const int wave = tid >> 5;
    const int lane = tid & 31;
    const int n    = lane & 15;      // N column within tile (B/C/D layout)
    const int hi   = lane >> 4;      // half-wave
    const int n0   = wave * 16;      // gate column base for this wave
    const int b0   = blockIdx.x * 16;
    const int dir  = blockIdx.y;

    const float* WihD  = Wih  + (size_t)dir * 128 * IN;
    const float* WhhD  = Whh  + (size_t)dir * 128 * 32;
    const float  bgate = bias[dir * 128 + n0 + n];

    // ---- build weight B-fragments once (K x 16 gate cols), kept in VGPRs ----
    // B layout: lane&15 = N, half j -> K = j + hi*16. Loads are contiguous ->
    // global_load_b128 + v_cvt_pk_bf16_f32, no masking.
    v16bf wBh;
#pragma unroll
    for (int j = 0; j < 16; ++j)
        wBh[j] = (__bf16)WhhD[(n0 + n) * 32 + j + hi * 16];

    constexpr int KC = (IN + 31) / 32;   // K chunks of 32 for the input GEMM
    v16bf wBx[KC];
#pragma unroll
    for (int kc = 0; kc < KC; ++kc) {
#pragma unroll
        for (int j = 0; j < 16; ++j) {
            int k = kc * 32 + j + hi * 16;
            // clamp address: padded-K positions meet a zero A operand, so the
            // (finite) clamped value is harmless -> unconditional load, no branch
            int kcl = (k < IN) ? k : (IN - 1);
            wBx[kc][j] = (__bf16)WihD[(n0 + n) * IN + kcl];
        }
    }

    // ---- zero h state ----
    for (int idx = tid; idx < 16 * 40; idx += 256)
        (&hs[0][0])[idx] = (__bf16)0.f;

    // each thread owns two ADJACENT cells (b, 2j) (b, 2j+1): paired LDS reads,
    // packed b32 stores. Cell state stays in registers.
    const int br = tid >> 4;          // batch row 0..15
    const int j2 = (tid & 15) * 2;    // even hidden column
    float c0 = 0.f, c1 = 0.f;
    float h0 = 0.f, h1 = 0.f;

    // per-thread streaming pointers (replaces per-step 64-bit address math)
    const long  tstep = dir ? -1 : 1;
    const long  t0    = dir ? (Tt - 1) : 0;
    const long  xrow  = (long)Bb * IN;            // elements per timestep of x
    const long  yrow  = (long)Bb * 64;            // elements per timestep of y
    const float*  xf  = (const float*)xin  + (t0 * (long)Bb + b0 + (tid >> 2)) * 4 + (tid & 3);
    const __bf16* xb  = (const __bf16*)xin + (t0 * (long)Bb + b0 + (tid >> 4)) * 64 + (tid & 15) * 4;
    __bf16*       yp  = WRITE_Y ? (yout + (t0 * (long)Bb + b0 + br) * 64 + dir * 32 + j2) : nullptr;

    for (int p = 0; p < Tt; ++p) {
        // ---- stage x[t] tile into LDS ----
        if (IN == 4) {
            if (tid < 64) {
                xsf[tid >> 2][tid & 3] = *xf;
                __builtin_prefetch(xf + tstep * xrow, 0, 0);  // speculative: OOB ok
                xf += tstep * xrow;
            }
        } else {
            // CDNA5 async LDS DMA (ASYNCcnt-tracked), 8B per lane
            unsigned int laddr = (unsigned int)(uintptr_t)&xsb[tid >> 4][(tid & 15) * 4];
            unsigned long long gaddr = (unsigned long long)(uintptr_t)xb;
            asm volatile("global_load_async_to_lds_b64 %0, %1, off"
                         :: "v"(laddr), "v"(gaddr) : "memory");
            __builtin_prefetch(xb + tstep * xrow, 0, 0);      // speculative: OOB ok
            xb += tstep * xrow;
            asm volatile("s_wait_asynccnt 0x0" ::: "memory");
        }
        __syncthreads();   // x staged; h from previous step visible

        // ---- build A fragments and run WMMAs ----
        const int m = n;   // A layout: lane&15 = M row
        v16bf ah;
#pragma unroll
        for (int j = 0; j < 16; ++j)
            ah[j] = hs[m][kmapA(j, hi)];

        v8f acc;
#pragma unroll
        for (int r = 0; r < 8; ++r) acc[r] = bgate;   // bias folded into C

        if (IN == 4) {
            // only half-wave 0, halves 0..3 are nonzero (K=0..3); build with
            // one aligned float4 LDS read + cndmask selects, no branches
            float4 xv = *(const float4*)&xsf[m][0];
            v16bf ax;
#pragma unroll
            for (int j = 0; j < 16; ++j) ax[j] = (__bf16)0.f;
            ax[0] = hi ? (__bf16)0.f : (__bf16)xv.x;
            ax[1] = hi ? (__bf16)0.f : (__bf16)xv.y;
            ax[2] = hi ? (__bf16)0.f : (__bf16)xv.z;
            ax[3] = hi ? (__bf16)0.f : (__bf16)xv.w;
            acc = __builtin_amdgcn_wmma_f32_16x16x32_bf16(
                      false, ax, false, wBx[0], (short)0, acc, false, false);
        } else {
#pragma unroll
            for (int kc = 0; kc < KC; ++kc) {
                v16bf ax;
#pragma unroll
                for (int j = 0; j < 16; ++j)
                    ax[j] = xsb[m][kc * 32 + kmapA(j, hi)];
                acc = __builtin_amdgcn_wmma_f32_16x16x32_bf16(
                          false, ax, false, wBx[kc], (short)0, acc, false, false);
            }
        }
        acc = __builtin_amdgcn_wmma_f32_16x16x32_bf16(
                  false, ah, false, wBh, (short)0, acc, false, false);

        // scatter D tile to LDS: lane&15 = N col, VGPR r -> row r + hi*8
#pragma unroll
        for (int r = 0; r < 8; ++r)
            zs[r + hi * 8][n0 + n] = acc[r];
        __syncthreads();   // z complete

        // ---- gate nonlinearities + state update (2 adjacent cells / thread) ----
        float zi0 = zs[br][j2],      zi1 = zs[br][j2 + 1];
        float zf0 = zs[br][32 + j2], zf1 = zs[br][33 + j2];
        float zg0 = zs[br][64 + j2], zg1 = zs[br][65 + j2];
        float zo0 = zs[br][96 + j2], zo1 = zs[br][97 + j2];
        c0 = sigm(zf0) * c0 + sigm(zi0) * tanhf(zg0);
        c1 = sigm(zf1) * c1 + sigm(zi1) * tanhf(zg1);
        h0 = sigm(zo0) * tanhf(c0);
        h1 = sigm(zo1) * tanhf(c1);

        PK2 pk;
        pk.h[0] = (__bf16)h0;
        pk.h[1] = (__bf16)h1;
        *(unsigned int*)&hs[br][j2] = pk.u;          // packed b32 store
        if (WRITE_Y) {
            *(unsigned int*)yp = pk.u;               // packed b32 store
            yp += tstep * yrow;
        }
        // next iteration's post-stage barrier orders the hs/zs hazards
        // (staging only touches xs*, which is quiescent here).
    }

    // final hidden features (hoisted out of the loop)
    PK2 pk;
    pk.h[0] = (__bf16)h0;
    pk.h[1] = (__bf16)h1;
    *(unsigned int*)&feat[(size_t)(b0 + br) * 192 + layer * 64 + dir * 32 + j2] = pk.u;
}

// =====================================================================
// Final FC: out[B,919] = feat[B,192] @ fc_w^T + fc_b, via 6 K-chunk WMMAs.
// grid = (B/16, 8); block = 256 (8 waves). wave w -> cols by*128 + 16w.
// Out-of-range columns: clamp load addresses (finite garbage accumulators),
// mask only the store -> zero exec-mask branching.
// =====================================================================
__global__ __launch_bounds__(256) void fc_kernel(
    const __bf16* __restrict__ feat,  // [B][192] bf16
    const float*  __restrict__ fcw,   // [919][192]
    const float*  __restrict__ fcb,   // [919]
    float*        __restrict__ out)   // [B][919]
{
    __shared__ alignas(16) __bf16 sA[16][200];    // staged feat tile (400B rows)

    const int tid  = threadIdx.x;
    const int wave = tid >> 5;
    const int lane = tid & 31;
    const int n    = lane & 15;
    const int hi   = lane >> 4;
    const int m0   = blockIdx.x * 16;
    const int col  = blockIdx.y * 128 + wave * 16 + n;
    const bool valid = col < Oo;
    const int colc = valid ? col : (Oo - 1);      // clamped, always-legal address

    // stage the 16x192 bf16 A tile (shared by all 8 waves)
    {
        int r = tid >> 4, cb = (tid & 15) * 12;
#pragma unroll
        for (int q = 0; q < 3; ++q)
            *(uint2*)&sA[r][cb + q * 4] =
                *(const uint2*)&feat[(size_t)(m0 + r) * 192 + cb + q * 4];
    }
    __syncthreads();

    v8f acc;
    const float bv = fcb[colc];
#pragma unroll
    for (int r = 0; r < 8; ++r) acc[r] = bv;

#pragma unroll
    for (int kc = 0; kc < 6; ++kc) {
        v16bf wb;
#pragma unroll
        for (int j = 0; j < 16; ++j)
            wb[j] = (__bf16)fcw[(size_t)colc * 192 + kc * 32 + j + hi * 16];
        v16bf aa;
#pragma unroll
        for (int j = 0; j < 16; ++j)
            aa[j] = sA[n][kc * 32 + kmapA(j, hi)];
        acc = __builtin_amdgcn_wmma_f32_16x16x32_bf16(
                  false, aa, false, wb, (short)0, acc, false, false);
    }

    if (valid) {
#pragma unroll
        for (int r = 0; r < 8; ++r)
            out[(size_t)(m0 + r + hi * 8) * Oo + col] = acc[r];
    }
}

// =====================================================================
extern "C" void kernel_launch(void* const* d_in, const int* in_sizes, int n_in,
                              void* d_out, int out_size, void* d_ws, size_t ws_size,
                              hipStream_t stream) {
    (void)in_sizes; (void)n_in; (void)out_size; (void)ws_size;

    const float* x     = (const float*)d_in[0];
    const float* Wih0  = (const float*)d_in[1];
    const float* Whh0  = (const float*)d_in[2];
    const float* b0    = (const float*)d_in[3];
    const float* Wih12 = (const float*)d_in[4];
    const float* Whh12 = (const float*)d_in[5];
    const float* b12   = (const float*)d_in[6];
    const float* fcw   = (const float*)d_in[7];
    const float* fcb   = (const float*)d_in[8];
    float* out = (float*)d_out;

    char* ws = (char*)d_ws;
    const size_t YB = (size_t)Tt * Bb * 64 * sizeof(__bf16);  // 131,072,000 B
    __bf16* bufA = (__bf16*)(ws);
    __bf16* bufB = (__bf16*)(ws + YB);
    __bf16* feat = (__bf16*)(ws + 2 * YB);

    dim3 grid(Bb / 16, 2), blk(256);
    // layer 0: fp32 x input (I=4), write y
    lstm_layer_kernel<4, true><<<grid, blk, 0, stream>>>(
        (const void*)x, Wih0, Whh0, b0, bufA, feat, 0);
    // layer 1: bf16 input (2H=64), write y
    lstm_layer_kernel<64, true><<<grid, blk, 0, stream>>>(
        (const void*)bufA, Wih12, Whh12, b12, bufB, feat, 1);
    // layer 2: y never consumed, only final hiddens
    lstm_layer_kernel<64, false><<<grid, blk, 0, stream>>>(
        (const void*)bufB, Wih12 + 2 * 128 * 64, Whh12 + 2 * 128 * 32,
        b12 + 2 * 128, nullptr, feat, 2);
    // final projection
    fc_kernel<<<dim3(Bb / 16, 8), blk, 0, stream>>>(feat, fcw, fcb, out);
}
